// BinaryTTN_69011534512336
// MI455X (gfx1250) — compile-verified, matrix operand order checked by reading, not run on hardware
//
#include <hip/hip_runtime.h>
#include <hip/hip_bf16.h>

// ---------- types ----------
typedef __attribute__((ext_vector_type(16))) __bf16        v16bf;
typedef __attribute__((ext_vector_type(8)))  float         v8f;
typedef __attribute__((ext_vector_type(4)))  unsigned int  uint4v;

// ---------- helpers ----------
// RNE f32 -> bf16
__device__ __forceinline__ unsigned int f2bf_rne(float f) {
    unsigned int x = __float_as_uint(f);
    return (x + 0x7FFFu + ((x >> 16) & 1u)) >> 16;
}
// pack two f32 into packed bf16 pair by truncation: one v_perm_b32
__device__ __forceinline__ unsigned int pack_bf16_trunc(float hi, float lo) {
    return __builtin_amdgcn_perm(__float_as_uint(hi), __float_as_uint(lo), 0x07060302u);
}
// duplicate low / high bf16 of a packed pair into both halves: one v_perm_b32
__device__ __forceinline__ unsigned int dup_lo16(unsigned int u) {
    return __builtin_amdgcn_perm(u, u, 0x01000100u);
}
__device__ __forceinline__ unsigned int dup_hi16(unsigned int u) {
    return __builtin_amdgcn_perm(u, u, 0x03020302u);
}
// packed bf16 multiply: v_pk_mul_bf16 (VOP3P) via inline asm
__device__ __forceinline__ unsigned int pk_mul_bf16(unsigned int a, unsigned int b) {
    unsigned int d;
    asm volatile("v_pk_mul_bf16 %0, %1, %2" : "=v"(d) : "v"(a), "v"(b));
    return d;
}
__device__ __forceinline__ float bf2f(unsigned short u) {
    return __uint_as_float(((unsigned int)u) << 16);
}

// =====================================================================
// Weight pre-conversion: fp32 -> packed bf16 (RNE), 8 floats per thread.
// count must be a multiple of 2048 (all layer weight tensors are).
// =====================================================================
__global__ void cvt_w_bf16(const float* __restrict__ src,
                           unsigned short* __restrict__ dst) {
    const long i = ((long)blockIdx.x * 256 + threadIdx.x) * 8;
    union { uint4v q; unsigned int u[4]; } o;
    #pragma unroll
    for (int k = 0; k < 4; ++k)
        o.u[k] = f2bf_rne(src[i + 2 * k]) | (f2bf_rne(src[i + 2 * k + 1]) << 16);
    *(uint4v*)(dst + i) = o.q;
}

// =====================================================================
// Layer 0: input x (n,2,64,64) f32, W0 (32,64,16,2,2) f32
// out[n,X,Y,b] = sum_{i,j} W0[X,Y,b,i,j] * x[n,i,2X,Y] * x[n,j,2X+1,Y]
// output bf16 (n,32,64,16). Memory-bound; VALU path.
// =====================================================================
__global__ void ttn_layer0(const float* __restrict__ x,
                           const float* __restrict__ W,
                           unsigned short* __restrict__ out) {
    __shared__ float sw[64];
    const int s = blockIdx.x;           // 0..2047 sites (32x64)
    const int X = s >> 6;
    const int Y = s & 63;
    const int t = threadIdx.x;
    if (t < 64) sw[t] = W[s * 64 + t];
    __syncthreads();

    for (int n = t; n < 2048; n += 256) {
        const long lbase = (((long)n * 2) * 64 + 2 * X) * 64 + Y;   // c=0, row 2X
        const float l0 = x[lbase];
        const float l1 = x[lbase + 4096];   // c=1
        const float r0 = x[lbase + 64];     // row 2X+1
        const float r1 = x[lbase + 64 + 4096];
        const float p00 = l0 * r0, p01 = l0 * r1, p10 = l1 * r0, p11 = l1 * r1;

        union { uint4v q; unsigned int u[4]; } o0, o1;
        #pragma unroll
        for (int k = 0; k < 8; ++k) {
            const float va = sw[(2 * k) * 4 + 0] * p00 + sw[(2 * k) * 4 + 1] * p01 +
                             sw[(2 * k) * 4 + 2] * p10 + sw[(2 * k) * 4 + 3] * p11;
            const float vb = sw[(2 * k + 1) * 4 + 0] * p00 + sw[(2 * k + 1) * 4 + 1] * p01 +
                             sw[(2 * k + 1) * 4 + 2] * p10 + sw[(2 * k + 1) * 4 + 3] * p11;
            const unsigned int packed = pack_bf16_trunc(vb, va);
            if (k < 4) o0.u[k] = packed; else o1.u[k - 4] = packed;
        }
        const long obase = (((long)n * 32 + X) * 64 + Y) * 16;
        *(uint4v*)(out + obase)     = o0.q;
        *(uint4v*)(out + obase + 8) = o1.q;
    }
}

// =====================================================================
// Layers 1..10 (in=16, bond=16): per site, GEMM [batch x 256] @ [256 x 16]
// via 8 accumulating v_wmma_f32_16x16x32_bf16. All dims are template
// constants so every stride folds into immediate offsets.
//   A[n, i*16+j] = L[n,i]*R[n,j]   (v_pk_mul_bf16 from packed bf16)
//   B[i*16+j, b] = W[site,b,i,j]   (pre-converted bf16, staged via LDS)
// Block: one site x 128 batches (8 wave32s, 16 rows each).
// =====================================================================
template <int H, int W, int HP, int WP, int ORIENT>
__global__ __launch_bounds__(256) void ttn_mid_t(
        const unsigned short* __restrict__ actIn,
        const unsigned short* __restrict__ Wbf,     // bf16 weights [site][b][i*16+j]
        unsigned short* __restrict__ actOut) {
    __shared__ __align__(16) unsigned short smW[4096];   // 8KB
    const int s = blockIdx.x;
    const int X = s / W, Y = s % W;                       // W is power of 2 -> shifts
    const int t = threadIdx.x;

    // stage bf16 weights into LDS: pure copy, 32B per thread
    {
        const unsigned short* wp = Wbf + (long)s * 4096 + t * 16;
        *(uint4v*)(smW + t * 16)     = *(const uint4v*)(wp);
        *(uint4v*)(smW + t * 16 + 8) = *(const uint4v*)(wp + 8);
    }
    __syncthreads();

    const int lane = t & 31;
    const int wv   = t >> 5;            // wave32
    const int half = lane >> 4;         // K sub-range selector (A/B) / M-offset (C)
    const int row  = lane & 15;         // A-matrix row M
    const int n0   = blockIdx.y * 128 + wv * 16;
    const int n    = n0 + row;

    const int xl = ORIENT ? X : 2 * X;
    const int yl = ORIENT ? 2 * Y : Y;
    const int xr = ORIENT ? X : 2 * X + 1;
    const int yr = ORIENT ? 2 * Y + 1 : Y;

    const unsigned short* lp = actIn + (((long)n * HP + xl) * WP + yl) * 16;
    const unsigned short* rp = actIn + (((long)n * HP + xr) * WP + yr) * 16;

    // L row as 8 packed-bf16 pairs; R: this lane's half (8 values = 4 pairs)
    unsigned int Lu[8], Ru[4];
    {
        union { uint4v q; unsigned int u[4]; } a, b, c;
        a.q = *(const uint4v*)(lp);
        b.q = *(const uint4v*)(lp + 8);
        c.q = *(const uint4v*)(rp + 8 * half);
        #pragma unroll
        for (int i = 0; i < 4; ++i) { Lu[i] = a.u[i]; Lu[4 + i] = b.u[i]; Ru[i] = c.u[i]; }
    }

    // preload all B chunks: 16x ds_load_b128, then dense WMMA chain
    uint4v Bq[16];
    {
        const int bbase = (lane & 15) * 256 + 8 * half;
        #pragma unroll
        for (int kk = 0; kk < 8; ++kk) {
            Bq[2 * kk]     = *(const uint4v*)(smW + bbase + kk * 32);
            Bq[2 * kk + 1] = *(const uint4v*)(smW + bbase + kk * 32 + 16);
        }
    }

    v8f c = {};
    #pragma unroll
    for (int kk = 0; kk < 8; ++kk) {    // K = 256 = 8 x 32
        // A chunk (16-bit A layout): elems 0..7 -> i=2kk, j=e+8*half
        //                            elems 8..15 -> i=2kk+1, j=(e-8)+8*half
        union { v16bf v; unsigned int u[8]; } A;
        const unsigned int li0 = dup_lo16(Lu[kk]);   // L[2kk]   in both halves
        const unsigned int li1 = dup_hi16(Lu[kk]);   // L[2kk+1] in both halves
        #pragma unroll
        for (int q = 0; q < 4; ++q) {
            A.u[q]     = pk_mul_bf16(li0, Ru[q]);
            A.u[4 + q] = pk_mul_bf16(li1, Ru[q]);
        }
        union { v16bf v; uint4v q[2]; } B;
        B.q[0] = Bq[2 * kk];
        B.q[1] = Bq[2 * kk + 1];

        c = __builtin_amdgcn_wmma_f32_16x16x32_bf16(false, A.v, false, B.v,
                                                    (short)0, c, false, false);
    }

    // C/D layout: lanes 0-15: M=vgpr, N=lane; lanes 16-31: M=8+vgpr, N=lane-16
    union { v8f v; float f[8]; } C;
    C.v = c;
    const int b = lane & 15;
    // base for row r=0; per-row stride H*W*16 is a compile-time constant, so
    // the 8 stores become base + immediate offsets.
    unsigned short* op = actOut + (((long)(n0 + 8 * half) * H + X) * W + Y) * 16 + b;
    #pragma unroll
    for (int r = 0; r < 8; ++r) {
        // store f32 high 16 bits = truncated bf16 (d16_hi store path)
        op[(long)r * H * W * 16] = (unsigned short)(__float_as_uint(C.f[r]) >> 16);
    }
}

// =====================================================================
// Layer 11: in (n,1,2,16) bf16, W11 (1,1,1,16,16) f32 -> out[n] f32
// =====================================================================
__global__ void ttn_layer11(const unsigned short* __restrict__ actIn,
                            const float* __restrict__ Wl,
                            float* __restrict__ out) {
    __shared__ float sw[256];
    const int t = threadIdx.x;
    sw[t] = Wl[t];
    __syncthreads();

    const int n = blockIdx.x * 256 + t;
    const unsigned short* lp = actIn + (long)n * 32;   // (n,0,0,:) then (n,0,1,:)
    union U16 { uint4v q; unsigned short us[8]; };
    float L[16], R[16];
    {
        U16 a, b, cdat, d;
        a.q = *(const uint4v*)(lp);
        b.q = *(const uint4v*)(lp + 8);
        cdat.q = *(const uint4v*)(lp + 16);
        d.q = *(const uint4v*)(lp + 24);
        #pragma unroll
        for (int i = 0; i < 8; ++i) {
            L[i] = bf2f(a.us[i]);    L[8 + i] = bf2f(b.us[i]);
            R[i] = bf2f(cdat.us[i]); R[8 + i] = bf2f(d.us[i]);
        }
    }
    float acc = 0.f;
    #pragma unroll
    for (int i = 0; i < 16; ++i) {
        float mi = 0.f;
        #pragma unroll
        for (int j = 0; j < 16; ++j) mi += sw[i * 16 + j] * R[j];
        acc += L[i] * mi;
    }
    out[n] = acc;
}

// =====================================================================
extern "C" void kernel_launch(void* const* d_in, const int* in_sizes, int n_in,
                              void* d_out, int out_size, void* d_ws, size_t ws_size,
                              hipStream_t stream) {
    (void)in_sizes; (void)n_in; (void)out_size; (void)ws_size;
    const float* x = (const float*)d_in[0];

    // workspace layout (ushort elements):
    //   bufA: [0, 67,108,864)                 layer0 out & even layers (134 MB)
    //   bufB: [67,108,864, 100,663,296)       odd layers (67 MB)
    //   bufW: [100,663,296, +8,380,416)       bf16 weights, layers 1..10 (16 MB)
    unsigned short* bufA = (unsigned short*)d_ws;
    unsigned short* bufB = bufA + (size_t)67108864;
    unsigned short* bufW = bufA + (size_t)100663296;

    // per-layer site counts (layers 1..10)
    const int sites[11] = {0, 1024, 512, 256, 128, 64, 32, 16, 8, 4, 2};

    // 1) pre-convert weights fp32 -> bf16 (RNE)
    {
        size_t off = 0;
        for (int i = 1; i <= 10; ++i) {
            const long cnt = (long)sites[i] * 4096;     // multiple of 2048
            cvt_w_bf16<<<dim3((unsigned)(cnt / 2048)), dim3(256), 0, stream>>>(
                (const float*)d_in[1 + i], bufW + off);
            off += (size_t)cnt;
        }
    }

    // 2) layer 0 (fp32 inputs, VALU)
    ttn_layer0<<<dim3(2048), dim3(256), 0, stream>>>(x, (const float*)d_in[1], bufA);

    // 3) layers 1..10: templated WMMA kernels, ping-pong bufA/bufB
    size_t woff = 0;
    unsigned short* cur = bufA;
    unsigned short* nxt = bufB;
    #define RUN_MID(i, H_, W_, HP_, WP_, OR_)                                          \
        do {                                                                           \
            ttn_mid_t<H_, W_, HP_, WP_, OR_>                                           \
                <<<dim3(H_ * W_, 16), dim3(256), 0, stream>>>(cur, bufW + woff, nxt);  \
            woff += (size_t)sites[i] * 4096;                                           \
            unsigned short* tmp = cur; cur = nxt; nxt = tmp;                           \
        } while (0)

    RUN_MID(1, 32, 32, 32, 64, 1);
    RUN_MID(2, 16, 32, 32, 32, 0);
    RUN_MID(3, 16, 16, 16, 32, 1);
    RUN_MID(4,  8, 16, 16, 16, 0);
    RUN_MID(5,  8,  8,  8, 16, 1);
    RUN_MID(6,  4,  8,  8,  8, 0);
    RUN_MID(7,  4,  4,  4,  8, 1);
    RUN_MID(8,  2,  4,  4,  4, 0);
    RUN_MID(9,  2,  2,  2,  4, 1);
    RUN_MID(10, 1,  2,  2,  2, 0);
    #undef RUN_MID

    // 4) layer 11 -> f32 output
    ttn_layer11<<<dim3(2048 / 256), dim3(256), 0, stream>>>(cur, (const float*)d_in[12],
                                                            (float*)d_out);
}
